// Seq2SeqLSTM_22986664968451
// MI455X (gfx1250) — matrix operand order repaired and checked by reading, not compiled, not run on hardware
//
#include <hip/hip_runtime.h>

// Problem constants (from reference)
#define L_STEPS 168
#define T_STEPS 24
#define B_DIM   512
#define IN_DIM  64
#define H_DIM   1024
#define G_DIM   (4 * H_DIM)   // 4096 gate width

typedef __attribute__((ext_vector_type(16))) _Float16 v16h;
typedef __attribute__((ext_vector_type(8)))  _Float16 v8h;
typedef __attribute__((ext_vector_type(8)))  float    v8f;

// ---------------------------------------------------------------------------
// fp32 -> fp16 conversion (with optional scale, used for x/10000)
// ---------------------------------------------------------------------------
__global__ void cvt_f32_f16(const float* __restrict__ src, _Float16* __restrict__ dst,
                            float scale, int n) {
  int i = blockIdx.x * blockDim.x + threadIdx.x;
  if (i < n) dst[i] = (_Float16)(src[i] * scale);
}

__device__ __forceinline__ float sigm(float x) { return 1.0f / (1.0f + expf(-x)); }

// Build a v16h A-fragment (16x32 f16, ISA 7.12.2): lane holds row m=l&15,
// halves 0-7 = K {0..7 | 8..15}, halves 8-15 = K {16..23 | 24..31} by l>>4.
__device__ __forceinline__ v16h load_afrag(const _Float16* p) {
  v8h lo = *(const v8h*)p;
  v8h hi = *(const v8h*)(p + 16);
  return __builtin_shufflevector(lo, hi, 0, 1, 2, 3, 4, 5, 6, 7,
                                 8, 9, 10, 11, 12, 13, 14, 15);
}
// B-fragment (32x16 f16): lane holds col n=l&15, K {0..7 | 8..15}, {16..23 | 24..31} by l>>4.
__device__ __forceinline__ v16h load_bfrag(const _Float16* p) {
  v8h lo = *(const v8h*)p;
  v8h hi = *(const v8h*)(p + 16);
  return __builtin_shufflevector(lo, hi, 0, 1, 2, 3, 4, 5, 6, 7,
                                 8, 9, 10, 11, 12, 13, 14, 15);
}

// ---------------------------------------------------------------------------
// Fully fused LSTM cell step:
//   z[b][g*H+j] = sum_k A0[b][k]*W0[g*H+j][k] (+ A1@W1^T) (+bias) (+addMat)
//   c_new = sig(z_f)*c + sig(z_i)*tanh(z_g);  h_new = sig(z_o)*tanh(c_new)
// Gate retiling: each wave computes one 16-wide j window for ALL 4 gates
// (N-tiles at g*H + j) x 2 M-tiles -> 8 v_wmma_f32_16x16x32_f16 accumulators.
// Since the C layout puts element (r + 8*(l>>4), l&15) at the same VGPR/lane
// in every gate accumulator, the LSTM pointwise runs entirely in registers.
// Workgroup: 256 threads = 8 waves (4 along M x 2 along j); tile 128M x 32j.
// ---------------------------------------------------------------------------
__global__ __launch_bounds__(256) void lstm_cell_fused(
    const _Float16* __restrict__ A0, const _Float16* __restrict__ W0, int K0,
    const _Float16* __restrict__ A1, const _Float16* __restrict__ W1, int K1,
    const float* __restrict__ addMat, const float* __restrict__ bias,
    float* __restrict__ c, _Float16* __restrict__ h16, float* __restrict__ h32) {
  const int lane   = threadIdx.x & 31;
  const int wave   = threadIdx.x >> 5;
  const int waveM  = wave & 3;   // 0..3 -> 32 rows each
  const int waveJ  = wave >> 2;  // 0..1 -> 16 cols each
  const int lane15 = lane & 15;
  const int laneHi = lane >> 4;

  const int mBase = blockIdx.y * 128 + waveM * 32;  // + mi*16
  const int jBase = blockIdx.x * 32 + waveJ * 16;   // hidden index window

  v8f acc[2][4];
#pragma unroll
  for (int mi = 0; mi < 2; ++mi)
#pragma unroll
    for (int g = 0; g < 4; ++g)
#pragma unroll
      for (int r = 0; r < 8; ++r) acc[mi][g][r] = 0.0f;

  for (int s = 0; s < 2; ++s) {
    const _Float16* A = s ? A1 : A0;
    const _Float16* W = s ? W1 : W0;
    const int K = s ? K1 : K0;
    if (A == nullptr) continue;
    const int kTiles = K >> 5;
    for (int kt = 0; kt < kTiles; ++kt) {
      const int kA = (kt << 5) + laneHi * 8;
      v16h afrag[2];
#pragma unroll
      for (int mi = 0; mi < 2; ++mi)
        afrag[mi] = load_afrag(A + (long)(mBase + mi * 16 + lane15) * K + kA);

      const int kB = (kt << 5) + laneHi * 8;
      v16h bfrag[4];
#pragma unroll
      for (int g = 0; g < 4; ++g)
        bfrag[g] =
            load_bfrag(W + (long)(g * H_DIM + jBase + lane15) * K + kB);

#pragma unroll
      for (int mi = 0; mi < 2; ++mi)
#pragma unroll
        for (int g = 0; g < 4; ++g)
          acc[mi][g] = __builtin_amdgcn_wmma_f32_16x16x32_f16(
              false, afrag[mi], false, bfrag[g], (short)0, acc[mi][g],
              false, false);
    }
  }

  // Fused LSTM epilogue, all in registers.
  const int j = jBase + lane15;
  float badd[4] = {0.0f, 0.0f, 0.0f, 0.0f};
  if (bias) {
#pragma unroll
    for (int g = 0; g < 4; ++g) badd[g] = bias[g * H_DIM + j];
  }
#pragma unroll
  for (int mi = 0; mi < 2; ++mi) {
#pragma unroll
    for (int r = 0; r < 8; ++r) {
      const int b = mBase + mi * 16 + laneHi * 8 + r;
      float vi = acc[mi][0][r] + badd[0];
      float vf = acc[mi][1][r] + badd[1];
      float vg = acc[mi][2][r] + badd[2];
      float vo = acc[mi][3][r] + badd[3];
      if (addMat) {
        const float* am = addMat + (long)b * G_DIM + j;
        vi += am[0];
        vf += am[H_DIM];
        vg += am[2 * H_DIM];
        vo += am[3 * H_DIM];
      }
      const long idx = (long)b * H_DIM + j;
      float cn = sigm(vf) * c[idx] + sigm(vi) * tanhf(vg);
      float hn = sigm(vo) * tanhf(cn);
      c[idx] = cn;
      h16[idx] = (_Float16)hn;
      if (h32) h32[idx] = hn;
    }
  }
}

// ---------------------------------------------------------------------------
// Plain WMMA GEMM (for one-time Ypre = y @ df_Wih^T + df_b precompute).
// Tile 128x128, 8 waves (2M x 4N), each wave 4x2 WMMA tiles.
// ---------------------------------------------------------------------------
__global__ __launch_bounds__(256) void gemm_f16(
    const _Float16* __restrict__ A, const _Float16* __restrict__ W, int K,
    const float* __restrict__ bias, float* __restrict__ Z, int N) {
  const int lane   = threadIdx.x & 31;
  const int wave   = threadIdx.x >> 5;
  const int waveM  = wave >> 2;
  const int waveN  = wave & 3;
  const int lane15 = lane & 15;
  const int laneHi = lane >> 4;
  const int mBase = blockIdx.y * 128 + waveM * 64;
  const int nBase = blockIdx.x * 128 + waveN * 32;

  v8f acc[4][2];
#pragma unroll
  for (int mi = 0; mi < 4; ++mi)
#pragma unroll
    for (int ni = 0; ni < 2; ++ni)
#pragma unroll
      for (int r = 0; r < 8; ++r) acc[mi][ni][r] = 0.0f;

  const int kTiles = K >> 5;
  for (int kt = 0; kt < kTiles; ++kt) {
    const int kA = (kt << 5) + laneHi * 8;
    v16h afrag[4];
#pragma unroll
    for (int mi = 0; mi < 4; ++mi)
      afrag[mi] = load_afrag(A + (long)(mBase + mi * 16 + lane15) * K + kA);
    const int kB = (kt << 5) + laneHi * 8;
    v16h bfrag[2];
#pragma unroll
    for (int ni = 0; ni < 2; ++ni)
      bfrag[ni] = load_bfrag(W + (long)(nBase + ni * 16 + lane15) * K + kB);
#pragma unroll
    for (int mi = 0; mi < 4; ++mi)
#pragma unroll
      for (int ni = 0; ni < 2; ++ni)
        acc[mi][ni] = __builtin_amdgcn_wmma_f32_16x16x32_f16(
            false, afrag[mi], false, bfrag[ni], (short)0, acc[mi][ni],
            false, false);
  }
#pragma unroll
  for (int mi = 0; mi < 4; ++mi)
#pragma unroll
    for (int ni = 0; ni < 2; ++ni) {
      const int n = nBase + ni * 16 + lane15;
      const float badd = bias ? bias[n] : 0.0f;
#pragma unroll
      for (int r = 0; r < 8; ++r) {
        const int m = mBase + mi * 16 + laneHi * 8 + r;
        Z[(long)m * N + n] = acc[mi][ni][r] + badd;
      }
    }
}

// ---------------------------------------------------------------------------
// Final projection: out[b] = dot(h[b, :], w)   (OUT == 1)
// ---------------------------------------------------------------------------
__global__ void dot_rows(const float* __restrict__ h, const float* __restrict__ w,
                         float* __restrict__ out) {
  __shared__ float red[256];
  int b = blockIdx.x;
  float s = 0.0f;
  for (int k = threadIdx.x; k < H_DIM; k += 256) s += h[(long)b * H_DIM + k] * w[k];
  red[threadIdx.x] = s;
  __syncthreads();
  for (int off = 128; off > 0; off >>= 1) {
    if (threadIdx.x < off) red[threadIdx.x] += red[threadIdx.x + off];
    __syncthreads();
  }
  if (threadIdx.x == 0) out[b] = red[0];
}

// ---------------------------------------------------------------------------
// Host orchestration
// ---------------------------------------------------------------------------
extern "C" void kernel_launch(void* const* d_in, const int* in_sizes, int n_in,
                              void* d_out, int out_size, void* d_ws, size_t ws_size,
                              hipStream_t stream) {
  (void)in_sizes; (void)n_in; (void)out_size; (void)ws_size;

  // Input order: x, then {ef,d1,d2,df,e1,e2} x {Wih,Whh,b}, then W_out
  const float* x      = (const float*)d_in[0];
  const float* efWih  = (const float*)d_in[1];
  const float* efWhh  = (const float*)d_in[2];
  const float* efB    = (const float*)d_in[3];
  const float* d1Wih  = (const float*)d_in[4];
  const float* d1Whh  = (const float*)d_in[5];
  const float* d1B    = (const float*)d_in[6];
  const float* d2Wih  = (const float*)d_in[7];
  const float* d2Whh  = (const float*)d_in[8];
  const float* d2B    = (const float*)d_in[9];
  const float* dfWih  = (const float*)d_in[10];
  const float* dfWhh  = (const float*)d_in[11];
  const float* dfB    = (const float*)d_in[12];
  const float* e1Wih  = (const float*)d_in[13];
  const float* e1Whh  = (const float*)d_in[14];
  const float* e1B    = (const float*)d_in[15];
  const float* e2Wih  = (const float*)d_in[16];
  const float* e2Whh  = (const float*)d_in[17];
  const float* e2B    = (const float*)d_in[18];
  const float* Wout   = (const float*)d_in[19];
  float* out = (float*)d_out;

  // Workspace carve-out (256B aligned), ~120 MB total
  size_t off = 0;
  auto alloc = [&](size_t bytes) -> void* {
    void* p = (char*)d_ws + off;
    off += (bytes + 255) & ~(size_t)255;
    return p;
  };
  const size_t GH = (size_t)G_DIM * H_DIM;   // 4096*1024
  const size_t GI = (size_t)G_DIM * IN_DIM;  // 4096*64
  const size_t BH = (size_t)B_DIM * H_DIM;   // 512*1024
  const size_t BG = (size_t)B_DIM * G_DIM;   // 512*4096

  _Float16* wEfIh = (_Float16*)alloc(GI * 2);
  _Float16* wEfHh = (_Float16*)alloc(GH * 2);
  _Float16* wD1Ih = (_Float16*)alloc(GH * 2);
  _Float16* wD1Hh = (_Float16*)alloc(GH * 2);
  _Float16* wD2Ih = (_Float16*)alloc(GH * 2);
  _Float16* wD2Hh = (_Float16*)alloc(GH * 2);
  _Float16* wDfIh = (_Float16*)alloc(GH * 2);
  _Float16* wDfHh = (_Float16*)alloc(GH * 2);
  _Float16* wE1Ih = (_Float16*)alloc(GH * 2);
  _Float16* wE1Hh = (_Float16*)alloc(GH * 2);
  _Float16* wE2Ih = (_Float16*)alloc(GH * 2);
  _Float16* wE2Hh = (_Float16*)alloc(GH * 2);
  _Float16* x16   = (_Float16*)alloc((size_t)L_STEPS * B_DIM * IN_DIM * 2);
  _Float16* y16   = (_Float16*)alloc(BH * 2);
  _Float16* hAll  = (_Float16*)alloc(3 * BH * 2);  // h0,h1,h2 (f16)
  float*    cAll  = (float*)   alloc(3 * BH * 4);  // c0,c1,c2 (f32)
  float*    h2f   = (float*)   alloc(BH * 4);      // f32 copy of h2 (decoder)
  float*    Ypre  = (float*)   alloc(BG * 4);      // y @ df_Wih^T + df_b

  _Float16 *h0 = hAll, *h1 = hAll + BH, *h2 = hAll + 2 * BH;
  float    *c0 = cAll, *c1 = cAll + BH, *c2 = cAll + 2 * BH;

  // --- once per launch: convert weights & x to f16, zero states ---
  auto cvt = [&](const float* s, _Float16* d, size_t n, float scale) {
    cvt_f32_f16<<<(int)((n + 255) / 256), 256, 0, stream>>>(s, d, scale, (int)n);
  };
  cvt(efWih, wEfIh, GI, 1.0f);  cvt(efWhh, wEfHh, GH, 1.0f);
  cvt(d1Wih, wD1Ih, GH, 1.0f);  cvt(d1Whh, wD1Hh, GH, 1.0f);
  cvt(d2Wih, wD2Ih, GH, 1.0f);  cvt(d2Whh, wD2Hh, GH, 1.0f);
  cvt(dfWih, wDfIh, GH, 1.0f);  cvt(dfWhh, wDfHh, GH, 1.0f);
  cvt(e1Wih, wE1Ih, GH, 1.0f);  cvt(e1Whh, wE1Hh, GH, 1.0f);
  cvt(e2Wih, wE2Ih, GH, 1.0f);  cvt(e2Whh, wE2Hh, GH, 1.0f);
  cvt(x, x16, (size_t)L_STEPS * B_DIM * IN_DIM, 1.0f / 10000.0f);
  hipMemsetAsync(hAll, 0, 3 * BH * 2, stream);
  hipMemsetAsync(cAll, 0, 3 * BH * 4, stream);

  const dim3 gCell(H_DIM / 32, B_DIM / 128);  // 32 x 4 workgroups
  auto cell = [&](const _Float16* A0, const _Float16* W0, int K0,
                  const _Float16* A1, const _Float16* W1, int K1,
                  const float* addM, const float* bias, float* c,
                  _Float16* h, float* h32) {
    lstm_cell_fused<<<gCell, 256, 0, stream>>>(A0, W0, K0, A1, W1, K1, addM,
                                               bias, c, h, h32);
  };

  // --- encoder: ef -> d1 -> d2, 168 steps, fully fused cells ---
  for (int t = 0; t < L_STEPS; ++t) {
    const _Float16* xt = x16 + (size_t)t * B_DIM * IN_DIM;
    cell(h0, wEfHh, H_DIM, xt, wEfIh, IN_DIM, nullptr, efB, c0, h0, nullptr);
    cell(h1, wD1Hh, H_DIM, h0, wD1Ih, H_DIM, nullptr, d1B, c1, h1, nullptr);
    cell(h2, wD2Hh, H_DIM, h1, wD2Ih, H_DIM, nullptr, d2B, c2, h2, nullptr);
  }

  // --- snapshot y = final encoder h2; precompute Ypre = y @ df_Wih^T + b ---
  hipMemcpyAsync(y16, h2, BH * 2, hipMemcpyDeviceToDevice, stream);
  gemm_f16<<<dim3(G_DIM / 128, B_DIM / 128), 256, 0, stream>>>(
      y16, wDfIh, H_DIM, dfB, Ypre, G_DIM);

  // --- decoder: df -> e1 -> e2, 24 steps; project h2 each step ---
  for (int t = 0; t < T_STEPS; ++t) {
    cell(h0, wDfHh, H_DIM, nullptr, nullptr, 0, Ypre, nullptr, c0, h0, nullptr);
    cell(h1, wE1Hh, H_DIM, h0, wE1Ih, H_DIM, nullptr, e1B, c1, h1, nullptr);
    cell(h2, wE2Hh, H_DIM, h1, wE2Ih, H_DIM, nullptr, e2B, c2, h2, h2f);
    dot_rows<<<B_DIM, 256, 0, stream>>>(h2f, Wout, out + (size_t)t * B_DIM);
  }
}
